// FencingModel_2697239461892
// MI455X (gfx1250) — matrix operand order repaired and checked by reading, not compiled, not run on hardware
//
#include <hip/hip_runtime.h>
#include <hip/hip_bf16.h>

typedef __attribute__((ext_vector_type(8)))  float   v8f;
typedef __attribute__((ext_vector_type(8)))  __bf16  v8bf;
typedef __attribute__((ext_vector_type(16))) __bf16  v16bf;

#define T_STEPS 512
#define IN_F    136
#define HID     128
#define KX_PAD  168   // x / w_ih LDS row pitch in halfs (K padded to 160 + 8 anti-bank-conflict)
#define KH_PAD  136   // h / w_hh LDS row pitch in halfs (128 + 8)
#define XCHUNKS 9     // ceil(16*136 / 256) per-thread pipelined x elements

// LDS layout (units: __bf16 elements)
#define WIH_OFF 0
#define WIH_ELEMS (512 * KX_PAD)              // 86016
#define WHH_OFF  (WIH_OFF + WIH_ELEMS)        // 86016
#define WHH_ELEMS (512 * KH_PAD)              // 69632
#define X_OFF    (WHH_OFF + WHH_ELEMS)        // 155648
#define X_ELEMS  (16 * KX_PAD)                // 2688
#define H_OFF    (X_OFF + X_ELEMS)            // 158336
#define H_ELEMS  (16 * KH_PAD)                // 2176
#define SMEM_ELEMS (H_OFF + H_ELEMS)          // 160512 halfs = 321024 B  (<= 320KB WGP LDS)

static __device__ inline v8f wmma_bf16(v16bf a, v16bf b, v8f c) {
    // (neg_a, A, neg_b, B, c_mod, C, reuse_a, reuse_b)
    return __builtin_amdgcn_wmma_f32_16x16x32_bf16(false, a, false, b, (short)0, c, false, false);
}

// A-fragment (16xK tile, M rows in lanes): lane L -> row L&15,
// K-chunks {kb..kb+7, kb+16..kb+23} with kb = k0 + 8*(L>=16)   (ISA 7.12.2 16-bit A layout)
static __device__ inline v16bf load_a(const __bf16* base, int pitch, int lane, int k0) {
    int m  = lane & 15;
    int kb = k0 + ((lane >> 4) << 3);
    const __bf16* p = base + m * pitch + kb;
    v8bf lo = *(const v8bf*)(p);
    v8bf hi = *(const v8bf*)(p + 16);
    return __builtin_shufflevector(lo, hi, 0,1,2,3,4,5,6,7,8,9,10,11,12,13,14,15);
}

// B-fragment (Kx16 tile = 16 rows of the weight matrix):
// lane L -> weight row n0+(L&15), K-chunks {kb..kb+7, kb+16..kb+23} with kb = k0 + 8*(L>=16)
static __device__ inline v16bf load_b(const __bf16* wbase, int pitch, int lane, int n0, int k0) {
    int n  = n0 + (lane & 15);
    int kb = k0 + ((lane >> 4) << 3);
    const __bf16* p = wbase + n * pitch + kb;
    v8bf lo = *(const v8bf*)(p);
    v8bf hi = *(const v8bf*)(p + 16);
    return __builtin_shufflevector(lo, hi, 0,1,2,3,4,5,6,7,8,9,10,11,12,13,14,15);
}

static __device__ inline float fast_sigmoid(float x) {
    return 1.0f / (1.0f + __expf(-x));
}
static __device__ inline float fast_tanh(float x) {
    float ax = fabsf(x);
    float t  = 1.0f - 2.0f / (__expf(2.0f * ax) + 1.0f);
    return copysignf(t, x);
}

__global__ __launch_bounds__(256)
void fencing_lstm_fused(const float* __restrict__ frames,
                        const float* __restrict__ w_ih,
                        const float* __restrict__ w_hh,
                        const float* __restrict__ w_fc,
                        const float* __restrict__ b_fc,
                        float* __restrict__ out)
{
    extern __shared__ __bf16 smem[];
    __bf16* wih = smem + WIH_OFF;   // [512][KX_PAD] bf16, gate rows reordered
    __bf16* whh = smem + WHH_OFF;   // [512][KH_PAD] bf16, gate rows reordered
    __bf16* xt  = smem + X_OFF;     // [16][KX_PAD] current frame tile
    __bf16* hb  = smem + H_OFF;     // [16][KH_PAD] hidden state

    const int tid  = threadIdx.x;
    const int wave = tid >> 5;      // 8 waves; wave w owns hidden cols [16w, 16w+15]
    const int lane = tid & 31;
    const int b0   = blockIdx.x * 16;

    // ---- Stage weights: fp32 -> bf16, gate-row reorder, zero padding ----
    // LDS row r: block b=r>>6 (hidden col block), gate g=(r>>4)&3, j=r&15
    // original row = g*128 + b*16 + j
    for (int idx = tid; idx < WIH_ELEMS; idx += 256) {
        int r = idx / KX_PAD, k = idx % KX_PAD;
        int src = ((r >> 4) & 3) * HID + (r >> 6) * 16 + (r & 15);
        float v = (k < IN_F) ? w_ih[src * IN_F + k] : 0.0f;
        wih[idx] = (__bf16)v;
    }
    for (int idx = tid; idx < WHH_ELEMS; idx += 256) {
        int r = idx / KH_PAD, k = idx % KH_PAD;
        int src = ((r >> 4) & 3) * HID + (r >> 6) * 16 + (r & 15);
        float v = (k < HID) ? w_hh[src * HID + k] : 0.0f;
        whh[idx] = (__bf16)v;
    }
    for (int idx = tid; idx < X_ELEMS; idx += 256) xt[idx] = (__bf16)0.0f;
    for (int idx = tid; idx < H_ELEMS; idx += 256) hb[idx] = (__bf16)0.0f;

    // ---- static per-thread x-pipeline offsets (no div/mod in hot loop) ----
    bool xvalid[XCHUNKS];
    long xgb[XCHUNKS];   // global element offset of this thread's element, at t=0
    int  xlb[XCHUNKS];   // LDS element offset
    #pragma unroll
    for (int j = 0; j < XCHUNKS; ++j) {
        int idx = tid + 256 * j;
        xvalid[j] = idx < 16 * IN_F;
        int r = idx / IN_F, k = idx % IN_F;
        if (!xvalid[j]) { r = 0; k = 0; }
        xgb[j] = (long)(b0 + r) * T_STEPS * IN_F + k;   // + t*IN_F per step
        xlb[j] = r * KX_PAD + k;
    }

    // ---- prologue: stage x(0) ----
    float xr[XCHUNKS];
    #pragma unroll
    for (int j = 0; j < XCHUNKS; ++j)
        if (xvalid[j]) xr[j] = frames[xgb[j]];
    #pragma unroll
    for (int j = 0; j < XCHUNKS; ++j)
        if (xvalid[j]) xt[xlb[j]] = (__bf16)xr[j];
    __syncthreads();

    const int n0 = wave * 64;            // this wave's reordered gate-row base
    v8f c_state = {0.f,0.f,0.f,0.f,0.f,0.f,0.f,0.f};

    for (int t = 0; t < T_STEPS; ++t) {
        // ---- issue next timestep's frame loads (latency hidden by GEMMs) ----
        const bool has_next = (t + 1) < T_STEPS;
        #pragma unroll
        for (int j = 0; j < XCHUNKS; ++j)
            if (has_next && xvalid[j]) xr[j] = frames[xgb[j] + (long)(t + 1) * IN_F];
        if (t + 2 < T_STEPS && tid < 16)
            __builtin_prefetch(&frames[(long)(b0 + tid) * T_STEPS * IN_F + (long)(t + 2) * IN_F], 0, 1);

        // ---- gates[16, 64] for this wave = x @ w_ih.T + h @ w_hh.T (reordered) ----
        v8f acc_i = {0.f,0.f,0.f,0.f,0.f,0.f,0.f,0.f};
        v8f acc_f = acc_i, acc_g = acc_i, acc_o = acc_i;

        #pragma unroll
        for (int kk = 0; kk < 5; ++kk) {
            v16bf a = load_a(xt, KX_PAD, lane, kk * 32);
            acc_i = wmma_bf16(a, load_b(wih, KX_PAD, lane, n0 +  0, kk * 32), acc_i);
            acc_f = wmma_bf16(a, load_b(wih, KX_PAD, lane, n0 + 16, kk * 32), acc_f);
            acc_g = wmma_bf16(a, load_b(wih, KX_PAD, lane, n0 + 32, kk * 32), acc_g);
            acc_o = wmma_bf16(a, load_b(wih, KX_PAD, lane, n0 + 48, kk * 32), acc_o);
        }
        #pragma unroll
        for (int kk = 0; kk < 4; ++kk) {
            v16bf a = load_a(hb, KH_PAD, lane, kk * 32);
            acc_i = wmma_bf16(a, load_b(whh, KH_PAD, lane, n0 +  0, kk * 32), acc_i);
            acc_f = wmma_bf16(a, load_b(whh, KH_PAD, lane, n0 + 16, kk * 32), acc_f);
            acc_g = wmma_bf16(a, load_b(whh, KH_PAD, lane, n0 + 32, kk * 32), acc_g);
            acc_o = wmma_bf16(a, load_b(whh, KH_PAD, lane, n0 + 48, kk * 32), acc_o);
        }

        // ---- activations + state update (c stays in registers per wave) ----
        v8f hnew;
        #pragma unroll
        for (int j = 0; j < 8; ++j) {
            float ig = fast_sigmoid(acc_i[j]);
            float fg = fast_sigmoid(acc_f[j]);
            float gg = fast_tanh(acc_g[j]);
            float og = fast_sigmoid(acc_o[j]);
            float c  = fg * c_state[j] + ig * gg;
            c_state[j] = c;
            hnew[j] = og * fast_tanh(c);
        }

        __syncthreads();   // all waves done reading old h and current x
        {
            // C/D layout: lane L holds col (L&15), rows j (L<16) or 8+j (L>=16)
            int col = (wave << 4) + (lane & 15);
            int r0  = (lane < 16) ? 0 : 8;
            #pragma unroll
            for (int j = 0; j < 8; ++j)
                hb[(r0 + j) * KH_PAD + col] = (__bf16)hnew[j];
        }
        // drain pipelined x(t+1) registers into the x tile
        #pragma unroll
        for (int j = 0; j < XCHUNKS; ++j)
            if (has_next && xvalid[j]) xt[xlb[j]] = (__bf16)xr[j];
        __syncthreads();   // new h and x visible to all waves
    }

    // ---- FC [16,128]x[128,3] + log_softmax, one thread per batch row ----
    if (tid < 16) {
        float logit[3];
        #pragma unroll
        for (int o = 0; o < 3; ++o) {
            float s = b_fc[o];
            for (int k = 0; k < HID; ++k)
                s += (float)hb[tid * KH_PAD + k] * w_fc[o * HID + k];
            logit[o] = s;
        }
        float m   = fmaxf(logit[0], fmaxf(logit[1], logit[2]));
        float se  = __expf(logit[0] - m) + __expf(logit[1] - m) + __expf(logit[2] - m);
        float lse = m + __logf(se);
        #pragma unroll
        for (int o = 0; o < 3; ++o)
            out[(long)(b0 + tid) * 3 + o] = logit[o] - lse;
    }
}

extern "C" void kernel_launch(void* const* d_in, const int* in_sizes, int n_in,
                              void* d_out, int out_size, void* d_ws, size_t ws_size,
                              hipStream_t stream) {
    const float* frames = (const float*)d_in[0];
    const float* w_ih   = (const float*)d_in[1];
    const float* w_hh   = (const float*)d_in[2];
    const float* w_fc   = (const float*)d_in[3];
    const float* b_fc   = (const float*)d_in[4];
    float* out = (float*)d_out;

    size_t smem_bytes = (size_t)SMEM_ELEMS * sizeof(__bf16);   // 321024 B
    fencing_lstm_fused<<<dim3(64), dim3(256), smem_bytes, stream>>>(
        frames, w_ih, w_hh, w_fc, b_fc, out);
}
